// SimpleGCN_76175539961908
// MI455X (gfx1250) — matrix operand order repaired and checked by reading, not compiled, not run on hardware
//
#include <hip/hip_runtime.h>
#include <hip/hip_bf16.h>

#define N_NODES 50000
#define N_EDGES 1600000
#define N_GRAPHS 1000
#define F_IN 92
#define WIDTH 128
#define F_OUT 64
#define MT 3125            // 50000 / 16, exact

typedef __attribute__((ext_vector_type(16))) __bf16 v16bf;
typedef __attribute__((ext_vector_type(8)))  float  v8f;

// ---------------------------------------------------------------- utilities

__global__ void k_zero2(int* __restrict__ a, int* __restrict__ b, int n) {
    int i = blockIdx.x * blockDim.x + threadIdx.x;
    if (i < n) { a[i] = 0; b[i] = 0; }
}

// edge weights + degree counting (int atomics: deterministic)
__global__ void k_edge(const float* __restrict__ bl, const int* __restrict__ src,
                       const int* __restrict__ dst, float* __restrict__ ew,
                       int* __restrict__ incnt, int* __restrict__ outcnt, int E) {
    int e = blockIdx.x * blockDim.x + threadIdx.x;
    if (e >= E) return;
    float x = bl[3*e], y = bl[3*e+1], z = bl[3*e+2];
    ew[e] = __expf(-(x*x + y*y + z*z) * (1.0f / 64.0f));   // lengthscale^2 = 64
    atomicAdd(&incnt[dst[e]], 1);
    atomicAdd(&outcnt[src[e]], 1);
}

// single-block exclusive scan over n counters -> row offsets + fill cursor
__global__ void k_scan(const int* __restrict__ cnt, int* __restrict__ ofs,
                       int* __restrict__ cur, int n) {
    __shared__ int sm[1024];
    __shared__ int carry;
    if (threadIdx.x == 0) carry = 0;
    __syncthreads();
    for (int base = 0; base < n; base += 1024) {
        int i = base + (int)threadIdx.x;
        int v = (i < n) ? cnt[i] : 0;
        sm[threadIdx.x] = v;
        __syncthreads();
        for (int off = 1; off < 1024; off <<= 1) {
            int t = (threadIdx.x >= (unsigned)off) ? sm[threadIdx.x - off] : 0;
            __syncthreads();
            sm[threadIdx.x] += t;
            __syncthreads();
        }
        int excl = sm[threadIdx.x] - v + carry;
        if (i < n) { ofs[i] = excl; cur[i] = excl; }
        __syncthreads();
        if (threadIdx.x == 1023) carry += sm[1023];
        __syncthreads();
    }
    if (threadIdx.x == 0) ofs[n] = carry;
}

__global__ void k_norm(const int* __restrict__ incnt, const int* __restrict__ outcnt,
                       float* __restrict__ nsrc, float* __restrict__ ndst, int n) {
    int i = blockIdx.x * blockDim.x + threadIdx.x;
    if (i >= n) return;
    nsrc[i] = rsqrtf((float)max(outcnt[i], 1));
    ndst[i] = rsqrtf((float)max(incnt[i], 1));
}

__global__ void k_fill(const int* __restrict__ dst, int* __restrict__ cur,
                       int* __restrict__ eid, int E) {
    int e = blockIdx.x * blockDim.x + threadIdx.x;
    if (e >= E) return;
    int p = atomicAdd(&cur[dst[e]], 1);
    eid[p] = e;
}

// W [K x N] f32 row-major  ->  Wt [N x KPAD] bf16 (zero-padded in K)
__global__ void k_convw(const float* __restrict__ W, __bf16* __restrict__ Wt,
                        int K, int N, int KPAD) {
    int idx = blockIdx.x * blockDim.x + threadIdx.x;
    if (idx >= N * KPAD) return;
    int n = idx / KPAD, k = idx % KPAD;
    float v = (k < K) ? W[k * N + n] : 0.0f;
    Wt[idx] = (__bf16)v;
}

// ---------------------------------------------------------------- WMMA GEMM
// Out[50000 x NOUT] = epilogue(A[50000 x KA] @ Wt^T), one wave per 16x64 strip.
template<int KA, int KPAD, int NOUT, bool RELU, bool HAS_BIAS, bool HAS_SCALE>
__global__ __launch_bounds__(256) void k_gemm(
    const float* __restrict__ A, const __bf16* __restrict__ Wt,
    const float* __restrict__ bias, const float* __restrict__ rowscale,
    float* __restrict__ Out)
{
    constexpr int NSTRIPS = NOUT / 64;
    const int lane = threadIdx.x & 31;
    const int wid  = (blockIdx.x * blockDim.x + threadIdx.x) >> 5;
    if (wid >= MT * NSTRIPS) return;          // wave-uniform exit: EXEC stays all-1s
    const int mt   = wid / NSTRIPS;
    const int nb   = (wid % NSTRIPS) * 64;
    const int m0   = mt * 16;
    const int half = lane >> 4;
    const int r    = lane & 15;
    const float* Arow = A + (size_t)(m0 + r) * KA;

    v8f c0 = {}, c1 = {}, c2 = {}, c3 = {};
    for (int k0 = 0; k0 < KPAD; k0 += 32) {
        // A fragment: row = lane, 16 contiguous Ks per lane-half (matches B)
        v16bf a;
        #pragma unroll
        for (int j = 0; j < 16; ++j) {
            int k = k0 + half * 16 + j;
            float v = (KA == KPAD || k < KA) ? Arow[k] : 0.0f;
            a[j] = (__bf16)v;
        }
        // B fragments: lane = column, 16 contiguous Ks per lane-half
        const int kh = k0 + half * 16;
        const __bf16* Wp = Wt + (size_t)(nb + r) * KPAD + kh;
        v16bf b0 = *(const v16bf*)(Wp);
        v16bf b1 = *(const v16bf*)(Wp + 16 * KPAD);
        v16bf b2 = *(const v16bf*)(Wp + 32 * KPAD);
        v16bf b3 = *(const v16bf*)(Wp + 48 * KPAD);
        c0 = __builtin_amdgcn_wmma_f32_16x16x32_bf16(false, a, false, b0, (short)0, c0, false, false);
        c1 = __builtin_amdgcn_wmma_f32_16x16x32_bf16(false, a, false, b1, (short)0, c1, false, false);
        c2 = __builtin_amdgcn_wmma_f32_16x16x32_bf16(false, a, false, b2, (short)0, c2, false, false);
        c3 = __builtin_amdgcn_wmma_f32_16x16x32_bf16(false, a, false, b3, (short)0, c3, false, false);
    }
    #pragma unroll
    for (int t = 0; t < 4; ++t) {
        v8f cc = (t == 0) ? c0 : (t == 1) ? c1 : (t == 2) ? c2 : c3;
        int col = nb + t * 16 + r;
        float badd = HAS_BIAS ? bias[col] : 0.0f;
        #pragma unroll
        for (int i = 0; i < 8; ++i) {
            int orow = m0 + half * 8 + i;      // C/D: element i -> M = half*8 + i
            float v = cc[i] + badd;
            if (RELU) v = fmaxf(v, 0.0f);
            if (HAS_SCALE) v *= rowscale[orow];
            Out[(size_t)orow * NOUT + col] = v;
        }
    }
}

// ------------------------------------------------------- CSR aggregations
// one wave32 per destination node; lane owns a float4 slice of 128 channels
__global__ __launch_bounds__(256) void k_agg1(
    const float* __restrict__ hs, const float* __restrict__ ew,
    const int* __restrict__ src, const int* __restrict__ eid,
    const int* __restrict__ ofs, const float* __restrict__ ndst,
    float* __restrict__ out)
{
    int wid  = (blockIdx.x * blockDim.x + threadIdx.x) >> 5;
    int lane = threadIdx.x & 31;
    if (wid >= N_NODES) return;
    int lo = ofs[wid], hi = ofs[wid + 1];
    float4 acc = make_float4(0.f, 0.f, 0.f, 0.f);
    for (int j = lo; j < hi; ++j) {
        int   e = eid[j];
        float w = ew[e];
        const float4 v = *(const float4*)(hs + (size_t)src[e] * WIDTH + lane * 4);
        acc.x = fmaf(v.x, w, acc.x);
        acc.y = fmaf(v.y, w, acc.y);
        acc.z = fmaf(v.z, w, acc.z);
        acc.w = fmaf(v.w, w, acc.w);
    }
    float nd = ndst[wid];
    *(float4*)(out + (size_t)wid * WIDTH + lane * 4) =
        make_float4(acc.x * nd, acc.y * nd, acc.z * nd, acc.w * nd);
}

// layer-2 aggregation over 64 channels; fuses norm_dst scale and b2 bias
__global__ __launch_bounds__(256) void k_agg2(
    const float* __restrict__ hs2, const float* __restrict__ ew,
    const int* __restrict__ src, const int* __restrict__ eid,
    const int* __restrict__ ofs, const float* __restrict__ ndst,
    const float* __restrict__ b2, float* __restrict__ out)
{
    int wid  = (blockIdx.x * blockDim.x + threadIdx.x) >> 5;
    int lane = threadIdx.x & 31;
    if (wid >= N_NODES) return;
    int lo = ofs[wid], hi = ofs[wid + 1];
    float2 acc = make_float2(0.f, 0.f);
    for (int j = lo; j < hi; ++j) {
        int   e = eid[j];
        float w = ew[e];
        const float2 v = *(const float2*)(hs2 + (size_t)src[e] * F_OUT + lane * 2);
        acc.x = fmaf(v.x, w, acc.x);
        acc.y = fmaf(v.y, w, acc.y);
    }
    float nd = ndst[wid];
    float* o = out + (size_t)wid * F_OUT + lane * 2;
    o[0] = acc.x * nd + b2[lane * 2 + 0];
    o[1] = acc.y * nd + b2[lane * 2 + 1];
}

// ---------------------------------------------------------------- pooling
__device__ __forceinline__ int lbound(const int* __restrict__ a, int n, int key) {
    int lo = 0, hi = n;
    while (lo < hi) { int mid = (lo + hi) >> 1; if (a[mid] < key) lo = mid + 1; else hi = mid; }
    return lo;
}

__global__ void k_pool(const float* __restrict__ xf, const int* __restrict__ gids,
                       float* __restrict__ out) {
    int g = blockIdx.x;                 // one block per graph, 64 threads = channels
    int c = threadIdx.x;
    int lo = lbound(gids, N_NODES, g);
    int hi = lbound(gids, N_NODES, g + 1);
    float s = 0.0f;
    for (int i = lo; i < hi; ++i) s += xf[(size_t)i * F_OUT + c];
    float cnt = (float)max(hi - lo, 1);
    out[(size_t)g * F_OUT + c] = s / cnt;
}

// ---------------------------------------------------------------- launcher
extern "C" void kernel_launch(void* const* d_in, const int* in_sizes, int n_in,
                              void* d_out, int out_size, void* d_ws, size_t ws_size,
                              hipStream_t stream) {
    const float* atom = (const float*)d_in[0];
    const float* bl   = (const float*)d_in[1];
    const int*   src  = (const int*)  d_in[2];
    const int*   dst  = (const int*)  d_in[3];
    const int*   gids = (const int*)  d_in[4];
    const float* W0   = (const float*)d_in[5];
    const float* b0   = (const float*)d_in[6];
    const float* W1   = (const float*)d_in[7];
    const float* b1   = (const float*)d_in[8];
    const float* W2   = (const float*)d_in[9];
    const float* b2   = (const float*)d_in[10];
    float* out = (float*)d_out;

    char* w = (char*)d_ws;
    auto alloc = [&](size_t bytes) -> void* {
        void* p = (void*)w;
        w += (bytes + 255) & ~(size_t)255;
        return p;
    };
    float*  ew     = (float*) alloc((size_t)N_EDGES * 4);
    int*    eid    = (int*)   alloc((size_t)N_EDGES * 4);
    int*    incnt  = (int*)   alloc((size_t)N_NODES * 4);
    int*    outcnt = (int*)   alloc((size_t)N_NODES * 4);
    int*    ofs    = (int*)   alloc((size_t)(N_NODES + 1) * 4);
    int*    cur    = (int*)   alloc((size_t)N_NODES * 4);
    float*  nsrc   = (float*) alloc((size_t)N_NODES * 4);
    float*  ndst   = (float*) alloc((size_t)N_NODES * 4);
    __bf16* Wt0    = (__bf16*)alloc((size_t)WIDTH * 96 * 2);
    __bf16* Wt1    = (__bf16*)alloc((size_t)WIDTH * WIDTH * 2);
    __bf16* Wt2    = (__bf16*)alloc((size_t)F_OUT * WIDTH * 2);
    float*  hs     = (float*) alloc((size_t)N_NODES * WIDTH * 4);  // also reused as xs
    float*  aggn   = (float*) alloc((size_t)N_NODES * WIDTH * 4);  // also reused as xfinal
    float*  hs2    = (float*) alloc((size_t)N_NODES * F_OUT * 4);

    const int TB = 256;
    const int nodeBlocks = (N_NODES + TB - 1) / TB;
    const int edgeBlocks = (N_EDGES + TB - 1) / TB;
    const int aggBlocks  = (N_NODES * 32 + TB - 1) / TB;

    // graph preprocessing: weights, degrees, CSR-by-dst
    k_zero2<<<nodeBlocks, TB, 0, stream>>>(incnt, outcnt, N_NODES);
    k_edge <<<edgeBlocks, TB, 0, stream>>>(bl, src, dst, ew, incnt, outcnt, N_EDGES);
    k_scan <<<1, 1024, 0, stream>>>(incnt, ofs, cur, N_NODES);
    k_norm <<<nodeBlocks, TB, 0, stream>>>(incnt, outcnt, nsrc, ndst, N_NODES);
    k_fill <<<edgeBlocks, TB, 0, stream>>>(dst, cur, eid, N_EDGES);

    // weight transpose/convert to bf16 fragment layout
    k_convw<<<(WIDTH *  96 + TB - 1) / TB, TB, 0, stream>>>(W0, Wt0,  F_IN, WIDTH,  96);
    k_convw<<<(WIDTH * 128 + TB - 1) / TB, TB, 0, stream>>>(W1, Wt1, WIDTH, WIDTH, 128);
    k_convw<<<(F_OUT * 128 + TB - 1) / TB, TB, 0, stream>>>(W2, Wt2, WIDTH, F_OUT, 128);

    // hs = (atom @ W0 + b0) * norm_src
    k_gemm<92, 96, 128, false, true, true>
        <<<(MT * 2 * 32 + TB - 1) / TB, TB, 0, stream>>>(atom, Wt0, b0, nsrc, hs);
    // aggn = segsum_dst(hs[src]*ew) * norm_dst
    k_agg1<<<aggBlocks, TB, 0, stream>>>(hs, ew, src, eid, ofs, ndst, aggn);
    // xs = relu(aggn @ W1 + b1) * norm_src   (xs reuses hs buffer)
    k_gemm<128, 128, 128, true, true, true>
        <<<(MT * 2 * 32 + TB - 1) / TB, TB, 0, stream>>>(aggn, Wt1, b1, nsrc, hs);
    // hs2 = xs @ W2
    k_gemm<128, 128, 64, false, false, false>
        <<<(MT * 32 + TB - 1) / TB, TB, 0, stream>>>(hs, Wt2, nullptr, nullptr, hs2);
    // xfinal = segsum_dst(hs2[src]*ew) * norm_dst + b2   (reuses aggn buffer)
    k_agg2<<<aggBlocks, TB, 0, stream>>>(hs2, ew, src, eid, ofs, ndst, b2, aggn);
    // per-graph average pooling (deterministic, binary search on sorted graph_ids)
    k_pool<<<N_GRAPHS, F_OUT, 0, stream>>>(aggn, gids, out);
}
